// RegionPurposalNetwork_63531156242518
// MI455X (gfx1250) — compile-verified, matrix operand order checked by reading, not run on hardware
//
#include <hip/hip_runtime.h>
#include <hip/hip_bf16.h>
#include <math.h>

typedef __attribute__((ext_vector_type(16))) __bf16 v16bf;
typedef __attribute__((ext_vector_type(8)))  float  v8f;
typedef __attribute__((ext_vector_type(4)))  int    v4i;

#define IMG_H 4200
#define IMG_W 3200
#define HF 84
#define WF 64
#define NB 8
#define ROW_F4 800   // 3200 floats per image row / 4

struct Bf16x16 { unsigned short s[16]; };

__device__ __forceinline__ unsigned short f2bf(float f) {
    unsigned int u = __builtin_bit_cast(unsigned int, f);
    unsigned int r = u + 0x7FFFu + ((u >> 16) & 1u);   // round-to-nearest-even
    return (unsigned short)(r >> 16);
}

#if __has_builtin(__builtin_amdgcn_global_load_async_to_lds_b128)
#define HAVE_ASYNC_LDS 1
typedef __attribute__((address_space(1))) v4i* gv4i_ptr;
typedef __attribute__((address_space(3))) v4i* lv4i_ptr;
#else
#define HAVE_ASYNC_LDS 0
#endif

#if __has_builtin(__builtin_amdgcn_s_wait_asynccnt)
#define WAIT_ASYNC(n) __builtin_amdgcn_s_wait_asynccnt(n)
#else
#define WAIT_ASYNC(n) asm volatile("s_wait_asynccnt %0" ::"i"(n) : "memory")
#endif

// Stage one 12.8KB image row into LDS. Async path: waves 0..4 each issue 5
// fully-coalesced 512B b128 copies (ASYNCcnt). Fallback: synchronous b128.
__device__ __forceinline__ void stage_row(const float4* __restrict__ src,
                                          float4* dst, int wid, int lane,
                                          int tid) {
#if HAVE_ASYNC_LDS
    if (wid < 5) {
        #pragma unroll
        for (int k = 0; k < 5; ++k) {
            const int i = wid * 160 + k * 32 + lane;
            __builtin_amdgcn_global_load_async_to_lds_b128(
                (gv4i_ptr)(void*)(src + i),
                (lv4i_ptr)(void*)(dst + i),
                0, 0);
        }
    }
#else
    for (int i = tid; i < ROW_F4; i += 256) dst[i] = src[i];
#endif
}

// ---------------------------------------------------------------------------
// Kernel A: 50x50 / stride-50 conv -> feats[8,2,84,64].
// Pure HBM stream of X (430MB ~= the whole roofline): one block per output
// row (hf,n); image rows double-buffered in LDS via async b128 copies;
// weights (20KB) in LDS; 4 threads per output column accumulate.
// ---------------------------------------------------------------------------
__global__ __launch_bounds__(256) void conv50_kernel(
    const float* __restrict__ X, const float* __restrict__ cw,
    const float* __restrict__ cb, float* __restrict__ feats) {
    __shared__ float  w0[2500];
    __shared__ float  w1[2500];
    __shared__ float4 rowbuf[2][ROW_F4];

    const int hf = blockIdx.x;
    const int n  = blockIdx.y;
    const int tid  = threadIdx.x;
    const int wid  = tid >> 5;
    const int lane = tid & 31;

    for (int i = tid; i < 2500; i += 256) { w0[i] = cw[i]; w1[i] = cw[2500 + i]; }

    const float* img = X + (size_t)n * IMG_H * IMG_W + (size_t)(hf * 50) * IMG_W;
    stage_row((const float4*)img, &rowbuf[0][0], wid, lane, tid);

    const int wf = tid >> 2;   // output column 0..63
    const int j  = tid & 3;    // column subset within the 50-wide patch row
    float a0 = 0.f, a1 = 0.f;

    for (int r = 0; r < 50; ++r) {
        if (r + 1 < 50)
            stage_row((const float4*)(img + (size_t)(r + 1) * IMG_W),
                      &rowbuf[(r + 1) & 1][0], wid, lane, tid);
#if HAVE_ASYNC_LDS
        if (wid < 5) {                 // wave-uniform
            if (r + 1 < 50) WAIT_ASYNC(5);   // row r's 5 copies retired (in-order)
            else            WAIT_ASYNC(0);
        }
#endif
        __syncthreads();               // staged row visible to all waves

        const float* row = (const float*)&rowbuf[r & 1][0] + wf * 50;
        const float* wr0 = w0 + r * 50;
        const float* wr1 = w1 + r * 50;
        #pragma unroll
        for (int k = 0; k < 13; ++k) {
            const int c = j + 4 * k;
            if (c < 50) {
                const float x = row[c];
                a0 = fmaf(x, wr0[c], a0);
                a1 = fmaf(x, wr1[c], a1);
            }
        }
        if (r + 1 < 50) __syncthreads();   // buffer reuse protection
    }

    a0 += __shfl_down(a0, 1, 4); a0 += __shfl_down(a0, 2, 4);
    a1 += __shfl_down(a1, 1, 4); a1 += __shfl_down(a1, 2, 4);
    if (j == 0) {
        feats[(((size_t)n * 2 + 0) * HF + hf) * WF + wf] = a0 + cb[0];
        feats[(((size_t)n * 2 + 1) * HF + hf) * WF + wf] = a1 + cb[1];
    }
}

// ---------------------------------------------------------------------------
// Kernel B: fused head. One block == one feature row (n,hf); 4 waves cover
// its 64 positions (16 each). LDS holds the zero-padded bf16 weight matrix
// (48x32), the padded feats rows and the combined bias vector, so the whole
// hot path (A/B builds, WMMA, spill) is branchless LDS traffic.
// Per wave: 3x v_wmma_f32_16x16x32_bf16 -> 16x48 f32 tile -> LDS ->
// softmax mask + anchors + clip -> out.
// ---------------------------------------------------------------------------
__global__ __launch_bounds__(128) void rpn_head_kernel(
    const float* __restrict__ feats,
    const float* __restrict__ box_w, const float* __restrict__ box_b,
    const float* __restrict__ cls_w, const float* __restrict__ cls_b,
    float* __restrict__ out) {
    __shared__ unsigned short Wb[48][32];    // bf16 weights, K>=18 zero-padded
    __shared__ unsigned short Fs[2][3][66];  // bf16 feats rows hf-1..hf+1, col-padded
    __shared__ float Bias[48];               // [box_b | cls_b | 0]
    __shared__ float tile[4][16][49];        // [wave][m][channel]

    const int tid  = threadIdx.x;
    const int wid  = tid >> 5;
    const int lane = tid & 31;
    const int n    = blockIdx.x / HF;
    const int hf   = blockIdx.x % HF;
    const int m    = lane & 15;              // position-in-tile (A) / column (B)
    const int half = lane >> 4;              // K-half selector
    const int wf   = wid * 16 + m;

    // combined [box|cls|pad] weight matrix, K = c*9 + dy*3 + dx
    for (int idx = tid; idx < 48 * 32; idx += 128) {
        const int N = idx >> 5, K = idx & 31;
        float wv = 0.f;
        if (K < 18) {
            if (N < 36)      wv = box_w[N * 18 + K];         // [36][2][3][3]
            else if (N < 38) wv = cls_w[(N - 36) * 18 + K];  // [2][2][3][3]
        }
        Wb[N][K] = f2bf(wv);
    }
    // combined bias vector
    if (tid < 48)
        Bias[tid] = (tid < 36) ? box_b[tid] : (tid < 38 ? cls_b[tid - 36] : 0.f);
    // padded feats slab: rows hf-1..hf+1, cols -1..64 (zeros outside)
    for (int idx = tid; idx < 2 * 3 * 66; idx += 128) {
        const int c = idx / 198, rem = idx % 198;
        const int rr = rem / 66, col = rem % 66;
        const int y = hf - 1 + rr, x = col - 1;
        float v = 0.f;
        if (y >= 0 && y < HF && x >= 0 && x < WF)
            v = feats[(((size_t)n * 2 + c) * HF + y) * WF + x];
        Fs[c][rr][col] = f2bf(v);
    }
    __syncthreads();

    // A: 16x32 bf16 (ISA 16-bit A striping: elem i -> K=(i<8?i:i+8)+8*half)
    Bf16x16 ab;
    #pragma unroll
    for (int i = 0; i < 16; ++i) {
        const int K  = (i < 8 ? i : i + 8) + half * 8;
        const int Kc = (K < 18) ? K : 0;
        const int c = Kc / 9, q = Kc % 9;
        const unsigned short us = Fs[c][q / 3][wf + (q % 3)];
        ab.s[i] = (K < 18) ? us : (unsigned short)0;
    }
    const v16bf A = __builtin_bit_cast(v16bf, ab);

    v8f acc[3];
    #pragma unroll
    for (int t = 0; t < 3; ++t) {
        Bf16x16 bb;
        #pragma unroll
        for (int i = 0; i < 16; ++i)
            bb.s[i] = Wb[t * 16 + m][(i < 8 ? i : i + 8) + half * 8];
        const v16bf B = __builtin_bit_cast(v16bf, bb);
        v8f c = {};
        c = __builtin_amdgcn_wmma_f32_16x16x32_bf16(
                false, A, false, B, (short)0, c, false, false);
        acc[t] = c;
    }

    // spill (+bias from LDS, branchless): D elem v -> row M = v + 8*half
    #pragma unroll
    for (int t = 0; t < 3; ++t) {
        const int N = t * 16 + m;
        const float bias = Bias[N];
        #pragma unroll
        for (int v = 0; v < 8; ++v)
            tile[wid][v + half * 8][N] = acc[t][v] + bias;
    }
    __syncthreads();

    // epilogue: softmax mask + anchors + clip; 18 channels per lane
    const float cls0 = tile[wid][m][36];
    const float cls1 = tile[wid][m][37];
    const float p1   = 1.f / (1.f + expf(cls0 - cls1));
    const float mk   = (p1 > 0.95f) ? 1.f : 0.f;
    const float cx = (wf + 0.5f) * 50.f;
    const float cy = (hf + 0.5f) * 50.f;
    const float SQ[3] = {0.70710678f, 1.0f, 1.41421356f};
    const size_t obase = ((size_t)((n * HF + hf) * WF + wf)) * 36;
    #pragma unroll
    for (int k = 0; k < 18; ++k) {
        const int ch = half * 18 + k;
        const int a = ch >> 2, d = ch & 3;
        const int si = a / 3, ri = a % 3;
        const float s  = (float)(16 << si);
        const float wa = s / SQ[ri];
        const float ha = s * SQ[ri];
        const float off = tile[wid][m][ch];
        float val;
        if (d == 0)      val = fminf(fmaxf(cx - 0.5f * wa + off, 0.f), (float)IMG_W);
        else if (d == 1) val = fminf(fmaxf(cy - 0.5f * ha + off, 0.f), (float)IMG_H);
        else if (d == 2) val = fminf(fmaxf(cx + 0.5f * wa + off, 0.f), (float)IMG_W);
        else             val = fminf(fmaxf(cy + 0.5f * ha + off, 0.f), (float)IMG_H);
        out[obase + ch] = val * mk;
    }
}

extern "C" void kernel_launch(void* const* d_in, const int* in_sizes, int n_in,
                              void* d_out, int out_size, void* d_ws, size_t ws_size,
                              hipStream_t stream) {
    const float* X      = (const float*)d_in[0];
    const float* conv_w = (const float*)d_in[1];
    const float* conv_b = (const float*)d_in[2];
    const float* box_w  = (const float*)d_in[3];
    const float* box_b  = (const float*)d_in[4];
    const float* cls_w  = (const float*)d_in[5];
    const float* cls_b  = (const float*)d_in[6];
    float* out   = (float*)d_out;
    float* feats = (float*)d_ws;    // 8*2*84*64 f32 = 336 KB scratch

    conv50_kernel<<<dim3(HF, NB), 256, 0, stream>>>(X, conv_w, conv_b, feats);
    rpn_head_kernel<<<dim3(NB * HF), 128, 0, stream>>>(feats, box_w, box_b,
                                                       cls_w, cls_b, out);
}
